// SimpleMoEModel_56203942035775
// MI455X (gfx1250) — matrix operand order repaired and checked by reading, not compiled, not run on hardware
//
#include <hip/hip_runtime.h>
#include <hip/hip_bf16.h>
#include <stdint.h>

// Problem sizes (match reference)
#define T_TOK 16384
#define H_DIM 1024
#define F_DIM 4096
#define E_NUM 8

typedef __attribute__((ext_vector_type(16))) __bf16 v16bf;
typedef __attribute__((ext_vector_type(8)))  float  v8f;

union FragAB { v16bf v; uint32_t u[8]; uint16_t s[16]; };

#if __has_builtin(__builtin_amdgcn_tensor_load_to_lds)
#define HAVE_TDM 1
#else
#define HAVE_TDM 0
#endif

// ---------- CDNA5 data-movement helpers ----------
// Per-lane async copy: 16B global -> LDS, tracked by ASYNCcnt.
__device__ __forceinline__ void async_copy_b128(void* lds_ptr, const void* gptr) {
  uint32_t lds_off = (uint32_t)(uintptr_t)lds_ptr;   // low 32 bits = LDS byte addr
  asm volatile("global_load_async_to_lds_b128 %0, %1, off"
               :: "v"(lds_off), "v"(gptr) : "memory");
}
__device__ __forceinline__ void wait_async() {
  asm volatile("s_wait_asynccnt 0x0" ::: "memory");
}
__device__ __forceinline__ void wait_tensor() {
#if __has_builtin(__builtin_amdgcn_s_wait_tensorcnt)
  __builtin_amdgcn_s_wait_tensorcnt(0);
#else
  asm volatile("s_wait_tensorcnt 0x0" ::: "memory");
#endif
}

// TDM: DMA a 2-D tile (tile_d0 contiguous 2B elems x tile_d1 rows) from a
// row-major bf16 tensor (row stride stride_d0 elems) into LDS, linearized
// row-after-row (rows land at lds_off + n*tile_d0*2).  D# per ISA 8.3/8.4.
__device__ __forceinline__ void tdm_load_2d(void* lds_ptr, const void* gaddr,
                                            uint32_t tensor_d0, uint32_t tensor_d1,
                                            uint32_t tile_d0, uint32_t tile_d1,
                                            uint32_t stride_d0) {
#if HAVE_TDM
  typedef __attribute__((ext_vector_type(4))) uint32_t v4u;
  typedef __attribute__((ext_vector_type(8))) int v8i;
  typedef __attribute__((ext_vector_type(4))) int v4i;
  uint64_t ga = (uint64_t)(uintptr_t)gaddr;
  v4u g0;
  g0.x = 1u;                                           // count=1, user D#
  g0.y = (uint32_t)(uintptr_t)lds_ptr;                 // lds_addr (bytes)
  g0.z = (uint32_t)ga;                                 // global_addr[31:0]
  g0.w = (uint32_t)((ga >> 32) & 0x01FFFFFFu) | (2u << 30);  // ga[56:32]|type=2
  v8i g1;
  g1[0] = (int)(1u << 16);                             // data_size=1 -> 2 bytes
  g1[1] = (int)((tensor_d0 & 0xFFFFu) << 16);          // abar=0 | tensor_dim0.lo
  g1[2] = (int)((tensor_d0 >> 16) | ((tensor_d1 & 0xFFFFu) << 16));
  g1[3] = (int)((tensor_d1 >> 16) | (tile_d0 << 16));  // tensor_dim1.hi|tile_dim0
  g1[4] = (int)tile_d1;                                // tile_dim1 | tile_dim2=0
  g1[5] = (int)stride_d0;                              // tensor_dim0_stride[31:0]
  g1[6] = 0;                                           // stride hi | dim1_stride
  g1[7] = 0;
  v4i z4 = {0, 0, 0, 0};
  v8i z8 = {0, 0, 0, 0, 0, 0, 0, 0};
  // 6-arg form (this toolchain): (g0, g1, g2, g3, g4, cpol)
  __builtin_amdgcn_tensor_load_to_lds(g0, g1, z4, z4, z8, 0);
#endif
}

// ---------- scalar helpers ----------
__device__ __forceinline__ uint16_t f2bf(float f) {
  uint32_t u = __float_as_uint(f);
  u += 0x7FFFu + ((u >> 16) & 1u);          // round-to-nearest-even
  return (uint16_t)(u >> 16);
}
__device__ __forceinline__ float bf2f(uint16_t h) {
  return __uint_as_float(((uint32_t)h) << 16);
}
__device__ __forceinline__ float gelu_tanh(float x) {
  float x3 = x * x * x;
  return 0.5f * x * (1.f + tanhf(0.7978845608028654f * (x + 0.044715f * x3)));
}

// ---------- WMMA fragment builders (contiguous-k order, same for A and B) ----------
// A (16x32, MxK): lane l holds row M=(l&15); koff=(l>=16)*16, K contiguous.
__device__ __forceinline__ v16bf load_a_frag(const uint16_t* lds, int mbase,
                                             int kbase, int stride, int lane) {
  FragAB f;
  int m = lane & 15, koff = (lane >> 4) * 16;
  const uint16_t* p = lds + (size_t)(mbase + m) * stride + kbase;
#pragma unroll
  for (int j = 0; j < 8; ++j) {
    f.u[j]     = *(const uint32_t*)(p + koff + 2 * j);
  }
  return f.v;
}
// B (32x16, KxN): lane l holds column N=(l&15); lanes 0-15 K=0..15,
// lanes 16-31 K=16..31. LDS stored [N][K] (K contiguous) -> 8 dword loads.
__device__ __forceinline__ v16bf load_b_fragT(const uint16_t* ldsT, int nbase,
                                              int kstride, int lane) {
  FragAB f;
  int n = lane & 15, khalf = (lane >> 4) * 16;
  const uint16_t* p = ldsT + (size_t)(nbase + n) * kstride + khalf;
#pragma unroll
  for (int j = 0; j < 8; ++j) f.u[j] = *(const uint32_t*)(p + 2 * j);
  return f.v;
}
// C/D (16x16 f32): lane l -> column N=(l&15); VGPR i -> row M = i + 8*(l>>4).

// ---------- kernel 0a: fp32 -> bf16 convert (no transpose) ----------
__global__ __launch_bounds__(256) void cvt_f32_bf16(const float* __restrict__ src,
                                                    uint16_t* __restrict__ dst, int n) {
  int stride = gridDim.x * blockDim.x * 4;
  for (int j = (blockIdx.x * blockDim.x + threadIdx.x) * 4; j < n; j += stride) {
    float4 f = *(const float4*)(src + j);
    uint32_t lo = (uint32_t)f2bf(f.x) | ((uint32_t)f2bf(f.y) << 16);
    uint32_t hi = (uint32_t)f2bf(f.z) | ((uint32_t)f2bf(f.w) << 16);
    uint2 o; o.x = lo; o.y = hi;
    *(uint2*)(dst + j) = o;
  }
}

// ---------- kernel 0b: fp32 [R][C] -> bf16 [C][R] tiled transpose ----------
__global__ __launch_bounds__(256) void transpose_cvt_bf16(const float* __restrict__ src,
                                                          uint16_t* __restrict__ dst,
                                                          int R, int C) {
  __shared__ uint16_t tile[64 * 65];         // padded stride vs bank conflicts
  int tid = threadIdx.x;
  int c0 = blockIdx.x * 64, r0 = blockIdx.y * 64;
  src += (size_t)blockIdx.z * R * C;
  dst += (size_t)blockIdx.z * R * C;
#pragma unroll
  for (int t = tid; t < 1024; t += 256) {    // read 64x64 f32, coalesced
    int r = t >> 4, c4 = (t & 15) * 4;
    float4 f = *(const float4*)(src + (size_t)(r0 + r) * C + c0 + c4);
    uint16_t* p = tile + r * 65 + c4;
    p[0] = f2bf(f.x); p[1] = f2bf(f.y); p[2] = f2bf(f.z); p[3] = f2bf(f.w);
  }
  __syncthreads();
#pragma unroll
  for (int t = tid; t < 512; t += 256) {     // write 64x64 bf16 transposed
    int co = t >> 3, r8 = (t & 7) * 8;
    uint16_t v[8];
#pragma unroll
    for (int j = 0; j < 8; ++j) v[j] = tile[(r8 + j) * 65 + co];
    *(uint4*)(dst + (size_t)(c0 + co) * R + r0 + r8) = *(const uint4*)v;
  }
}

// ---------- kernel 1: h = x @ attn_w + b  (bf16 WMMA, double-buffered) ----------
// awT is attn_w pre-transposed: [j][k] bf16, K contiguous.
__global__ __launch_bounds__(256) void attn_gemm(const uint16_t* __restrict__ xb,
                                                 const uint16_t* __restrict__ awT,
                                                 const float* __restrict__ bias,
                                                 uint16_t* __restrict__ hb) {
  __shared__ uint16_t lx[2 * 128 * 32];    // A tiles, row-major [m][k], x2
  __shared__ uint16_t lwT[2 * 128 * 32];   // B tiles, [n][k], x2
  int tid = threadIdx.x, lane = tid & 31, wave = tid >> 5;
  int wm = wave & 1, wn = wave >> 1;   // 2 M halves x 4 N strips
  int j0 = blockIdx.x * 128, t0 = blockIdx.y * 128;

  v8f zero = {0.f, 0.f, 0.f, 0.f, 0.f, 0.f, 0.f, 0.f};
  v8f acc[4][2];
#pragma unroll
  for (int mi = 0; mi < 4; ++mi)
#pragma unroll
    for (int ni = 0; ni < 2; ++ni) acc[mi][ni] = zero;

  // prologue: stage k-chunk 0 into buffer 0
#pragma unroll
  for (int c = tid; c < 512; c += 256) {
    int r = c >> 2, k8 = (c & 3) * 8;
    async_copy_b128(lx + r * 32 + k8, xb + (size_t)(t0 + r) * H_DIM + k8);
    async_copy_b128(lwT + r * 32 + k8, awT + (size_t)(j0 + r) * H_DIM + k8);
  }
  wait_async();
  __syncthreads();

  for (int kc = 0; kc < H_DIM; kc += 32) {
    int cb = (kc >> 5) & 1, nb = cb ^ 1;
    const uint16_t* lxc  = lx  + cb * (128 * 32);
    const uint16_t* lwc  = lwT + cb * (128 * 32);
    if (kc + 32 < H_DIM) {               // overlap DMA of next chunk with WMMA
#pragma unroll
      for (int c = tid; c < 512; c += 256) {
        int r = c >> 2, k8 = (c & 3) * 8;
        async_copy_b128(lx + nb * (128 * 32) + r * 32 + k8,
                        xb + (size_t)(t0 + r) * H_DIM + kc + 32 + k8);
        async_copy_b128(lwT + nb * (128 * 32) + r * 32 + k8,
                        awT + (size_t)(j0 + r) * H_DIM + kc + 32 + k8);
      }
    }
    v16bf bfrag[2];
#pragma unroll
    for (int ni = 0; ni < 2; ++ni)
      bfrag[ni] = load_b_fragT(lwc, wn * 32 + ni * 16, 32, lane);
#pragma unroll
    for (int mi = 0; mi < 4; ++mi) {
      v16bf a = load_a_frag(lxc, wm * 64 + mi * 16, 0, 32, lane);
#pragma unroll
      for (int ni = 0; ni < 2; ++ni)
        acc[mi][ni] = __builtin_amdgcn_wmma_f32_16x16x32_bf16(
            false, a, false, bfrag[ni], (short)0, acc[mi][ni], false, false);
    }
    wait_async();
    __syncthreads();
  }
  int n = lane & 15, hi = lane >> 4;
#pragma unroll
  for (int mi = 0; mi < 4; ++mi)
#pragma unroll
    for (int ni = 0; ni < 2; ++ni) {
      int col = j0 + wn * 32 + ni * 16 + n;
      float bv = bias[col];
#pragma unroll
      for (int i = 0; i < 8; ++i) {
        int row = t0 + wm * 64 + mi * 16 + hi * 8 + i;
        hb[(size_t)row * H_DIM + col] = f2bf(acc[mi][ni][i] + bv);
      }
    }
}

// ---------- kernel 2: router (scores -> softmax -> top-2) ----------
__global__ __launch_bounds__(256) void router(const uint16_t* __restrict__ hb,
                                              const float* __restrict__ rw,
                                              int* __restrict__ topi,
                                              float* __restrict__ topv) {
  int lane = threadIdx.x & 31, wave = threadIdx.x >> 5;
  int t = blockIdx.x * 8 + wave;
  float s[E_NUM] = {0.f, 0.f, 0.f, 0.f, 0.f, 0.f, 0.f, 0.f};
  for (int k = lane; k < H_DIM; k += 32) {
    float hv = bf2f(hb[(size_t)t * H_DIM + k]);
#pragma unroll
    for (int e = 0; e < E_NUM; ++e) s[e] += hv * rw[k * E_NUM + e];
  }
#pragma unroll
  for (int off = 16; off >= 1; off >>= 1)
#pragma unroll
    for (int e = 0; e < E_NUM; ++e) s[e] += __shfl_down(s[e], off, 32);
  if (lane == 0) {
    float m = s[0];
#pragma unroll
    for (int e = 1; e < E_NUM; ++e) m = fmaxf(m, s[e]);
    float p[E_NUM], sum = 0.f;
#pragma unroll
    for (int e = 0; e < E_NUM; ++e) { p[e] = __expf(s[e] - m); sum += p[e]; }
    float inv = 1.f / sum;
#pragma unroll
    for (int e = 0; e < E_NUM; ++e) p[e] *= inv;
    int i1 = 0; float v1 = p[0];
#pragma unroll
    for (int e = 1; e < E_NUM; ++e) if (p[e] > v1) { v1 = p[e]; i1 = e; }
    int i2 = -1; float v2 = -1.f;
#pragma unroll
    for (int e = 0; e < E_NUM; ++e)
      if (e != i1 && p[e] > v2) { v2 = p[e]; i2 = e; }
    topi[t * 2] = i1;  topv[t * 2] = v1;
    topi[t * 2 + 1] = i2;  topv[t * 2 + 1] = v2;
  }
}

// ---------- kernel 3: deterministic per-expert compaction ----------
__global__ __launch_bounds__(256) void build_lists(const int* __restrict__ topi,
                                                   const float* __restrict__ topv,
                                                   int* __restrict__ cnt,
                                                   int* __restrict__ idxb,
                                                   float* __restrict__ gvb) {
  __shared__ int sflag[256];
  __shared__ int sbase;
  int e = blockIdx.x, tid = threadIdx.x;
  if (tid == 0) sbase = 0;
  __syncthreads();
  for (int base = 0; base < T_TOK; base += 256) {
    int t = base + tid;
    int sel = 0; float g = 0.f;
    int a = topi[t * 2], b = topi[t * 2 + 1];
    if (a == e)      { sel = 1; g = topv[t * 2]; }
    else if (b == e) { sel = 1; g = topv[t * 2 + 1]; }
    sflag[tid] = sel;
    __syncthreads();
    for (int off = 1; off < 256; off <<= 1) {   // inclusive scan
      int v = (tid >= off) ? sflag[tid - off] : 0;
      __syncthreads();
      sflag[tid] += v;
      __syncthreads();
    }
    int pos = sbase + sflag[tid] - 1;
    if (sel) { idxb[e * T_TOK + pos] = t; gvb[e * T_TOK + pos] = g; }
    __syncthreads();
    if (tid == 255) sbase += sflag[255];
    __syncthreads();
  }
  if (tid == 255) cnt[e] = sbase;
}

// ---------- kernel 4: routed fused expert MLP (pipelined DMA) ----------
// w1T: [E][F][H] bf16 (pre-transposed), w2T: [E][H][F] bf16 (pre-transposed).
__global__ __launch_bounds__(256) void moe_expert(const uint16_t* __restrict__ hb,
                                                  const uint16_t* __restrict__ w1T,
                                                  const uint16_t* __restrict__ w2T,
                                                  const int* __restrict__ cnt,
                                                  const int* __restrict__ idxb,
                                                  const float* __restrict__ gvb,
                                                  float* __restrict__ out) {
  extern __shared__ uint16_t smem[];
  uint16_t* lh   = smem;                 // 32x1024 gathered h rows (64 KB)
  uint16_t* lact = lh + 32 * 1024;       // 32x128 gelu activations (8 KB)
  uint16_t* lw1T = lact + 32 * 128;      // 2 x [128 n][32 k] w1 chunks (16 KB)
  uint16_t* lw2T = lw1T + 2 * 128 * 32;  // [1024 n][32 k] w2 chunk (64 KB)
  __shared__ int   ltok[32];
  __shared__ float lgate[32];

  int tid = threadIdx.x, lane = tid & 31, wave = tid >> 5;
  int e = blockIdx.x >> 9;               // 512 tiles per expert
  int mt = blockIdx.x & 511;
  int rowStart = mt * 32;
  int count = cnt[e];
  if (rowStart >= count) return;         // uniform exit
  int nrows = min(32, count - rowStart);

  if (tid < 32) {
    if (tid < nrows) {
      ltok[tid]  = idxb[e * T_TOK + rowStart + tid];
      lgate[tid] = gvb[e * T_TOK + rowStart + tid];
    } else { ltok[tid] = 0; lgate[tid] = 0.f; }
  }
  __syncthreads();
#pragma unroll
  for (int c = tid; c < 4096; c += 256) {          // gather 32x1024 h rows
    int r = c >> 7, n8 = (c & 127) * 8;
    async_copy_b128(lh + r * 1024 + n8, hb + (size_t)ltok[r] * H_DIM + n8);
  }
  wait_async();
  __syncthreads();

  const uint16_t* w1e = w1T + (size_t)e * F_DIM * H_DIM;   // [F][H]
  const uint16_t* w2e = w2T + (size_t)e * H_DIM * F_DIM;   // [H][F]

  v8f zero = {0.f, 0.f, 0.f, 0.f, 0.f, 0.f, 0.f, 0.f};
  v8f acc2[2][8];
#pragma unroll
  for (int mi = 0; mi < 2; ++mi)
#pragma unroll
    for (int nt = 0; nt < 8; ++nt) acc2[mi][nt] = zero;

  for (int fc = 0; fc < F_DIM; fc += 128) {
#if HAVE_TDM
    // lw2T is idle during stage 1: launch the 64KB TDM for kk=0 now so the
    // tensor DMA overlaps the entire stage-1 WMMA work.
    if (wave == 0)
      tdm_load_2d(lw2T, w2e + fc, F_DIM, H_DIM, 32, H_DIM, F_DIM);
#endif
    if (fc + 128 < F_DIM)                 // speculative prefetch of next chunk
      __builtin_prefetch(w1e + (size_t)(fc + 128 + (tid >> 2)) * H_DIM, 0, 1);

    // ---- stage 1: act[32,128] = gelu(h[32,1024] @ w1[:, fc:fc+128]) ----
    v8f acc1[2]; acc1[0] = zero; acc1[1] = zero;
#pragma unroll
    for (int c = tid; c < 512; c += 256) {         // prologue: chunk 0 -> buf 0
      int n = c >> 2, k8 = (c & 3) * 8;
      async_copy_b128(lw1T + n * 32 + k8, w1e + (size_t)(fc + n) * H_DIM + k8);
    }
    wait_async();
    __syncthreads();
    for (int kc = 0; kc < H_DIM; kc += 32) {
      int cb = (kc >> 5) & 1, nb = cb ^ 1;
      const uint16_t* cur = lw1T + cb * (128 * 32);
      if (kc + 32 < H_DIM) {                       // DMA next chunk, overlapped
#pragma unroll
        for (int c = tid; c < 512; c += 256) {
          int n = c >> 2, k8 = (c & 3) * 8;
          async_copy_b128(lw1T + nb * (128 * 32) + n * 32 + k8,
                          w1e + (size_t)(fc + n) * H_DIM + kc + 32 + k8);
        }
      }
      v16bf b = load_b_fragT(cur, wave * 16, 32, lane);
#pragma unroll
      for (int mi = 0; mi < 2; ++mi) {
        v16bf a = load_a_frag(lh, mi * 16, kc, 1024, lane);
        acc1[mi] = __builtin_amdgcn_wmma_f32_16x16x32_bf16(
            false, a, false, b, (short)0, acc1[mi], false, false);
      }
      wait_async();
      __syncthreads();
    }
    {
      int n = lane & 15, hi = lane >> 4;
#pragma unroll
      for (int mi = 0; mi < 2; ++mi)
#pragma unroll
        for (int i = 0; i < 8; ++i) {
          int row = mi * 16 + hi * 8 + i;
          lact[row * 128 + wave * 16 + n] = f2bf(gelu_tanh(acc1[mi][i]));
        }
    }
    __syncthreads();
    // ---- stage 2: out[32,1024] += act[32,128] @ w2[fc:fc+128, :] ----
    for (int kk = 0; kk < 4; ++kk) {
#if HAVE_TDM
      if (wave == 0) wait_tensor();      // kk's tile staged (kk=0 during stage1)
      __syncthreads();
#else
#pragma unroll
      for (int c = tid; c < 4096; c += 256) {      // w2T rows 0..1023, k chunk
        int n = c >> 2, k8 = (c & 3) * 8;
        async_copy_b128(lw2T + n * 32 + k8,
                        w2e + (size_t)n * F_DIM + fc + kk * 32 + k8);
      }
      wait_async();
      __syncthreads();
#endif
      v16bf a[2];
#pragma unroll
      for (int mi = 0; mi < 2; ++mi)
        a[mi] = load_a_frag(lact, mi * 16, kk * 32, 128, lane);
#pragma unroll
      for (int nt = 0; nt < 8; ++nt) {
        v16bf b = load_b_fragT(lw2T, wave * 128 + nt * 16, 32, lane);
#pragma unroll
        for (int mi = 0; mi < 2; ++mi)
          acc2[mi][nt] = __builtin_amdgcn_wmma_f32_16x16x32_bf16(
              false, a[mi], false, b, (short)0, acc2[mi][nt], false, false);
      }
      __syncthreads();                   // all waves done reading lw2T
#if HAVE_TDM
      if (kk < 3 && wave == 0)           // DMA next 64KB tile
        tdm_load_2d(lw2T, w2e + fc + (kk + 1) * 32,
                    F_DIM, H_DIM, 32, H_DIM, F_DIM);
#endif
    }
  }
  // ---- epilogue: gate-scale + scatter-add (2 commutative adds/elem) ----
  int n = lane & 15, hi = lane >> 4;
#pragma unroll
  for (int mi = 0; mi < 2; ++mi)
#pragma unroll
    for (int nt = 0; nt < 8; ++nt)
#pragma unroll
      for (int i = 0; i < 8; ++i) {
        int row = mi * 16 + hi * 8 + i;
        if (row < nrows) {
          int col = wave * 128 + nt * 16 + n;
          atomicAdd(out + (size_t)ltok[row] * H_DIM + col,
                    lgate[row] * acc2[mi][nt][i]);
        }
      }
}

// ---------- workspace layout (bytes) ----------
#define OFF_XB   ((size_t)0)                        // T*H bf16
#define OFF_AWT  (OFF_XB  + (size_t)T_TOK * H_DIM * 2)
#define OFF_HB   (OFF_AWT + (size_t)H_DIM * H_DIM * 2)
#define OFF_W1T  (OFF_HB  + (size_t)T_TOK * H_DIM * 2)
#define OFF_W2T  (OFF_W1T + (size_t)E_NUM * H_DIM * F_DIM * 2)
#define OFF_CNT  (OFF_W2T + (size_t)E_NUM * F_DIM * H_DIM * 2)
#define OFF_IDX  (OFF_CNT + 256)
#define OFF_GV   (OFF_IDX + (size_t)E_NUM * T_TOK * 4)
#define OFF_TOPI (OFF_GV  + (size_t)E_NUM * T_TOK * 4)
#define OFF_TOPV (OFF_TOPI + (size_t)T_TOK * 2 * 4)

extern "C" void kernel_launch(void* const* d_in, const int* in_sizes, int n_in,
                              void* d_out, int out_size, void* d_ws, size_t ws_size,
                              hipStream_t stream) {
  const float* x        = (const float*)d_in[0];
  const float* attn_w   = (const float*)d_in[1];
  const float* attn_b   = (const float*)d_in[2];
  const float* router_w = (const float*)d_in[3];
  const float* w1       = (const float*)d_in[4];
  const float* w2       = (const float*)d_in[5];
  char* ws = (char*)d_ws;
  uint16_t* xb   = (uint16_t*)(ws + OFF_XB);
  uint16_t* awT  = (uint16_t*)(ws + OFF_AWT);
  uint16_t* hb   = (uint16_t*)(ws + OFF_HB);
  uint16_t* w1t  = (uint16_t*)(ws + OFF_W1T);
  uint16_t* w2t  = (uint16_t*)(ws + OFF_W2T);
  int*      cntb = (int*)(ws + OFF_CNT);
  int*      idxb = (int*)(ws + OFF_IDX);
  float*    gvb  = (float*)(ws + OFF_GV);
  int*      topi = (int*)(ws + OFF_TOPI);
  float*    topv = (float*)(ws + OFF_TOPV);
  float*    out  = (float*)d_out;

  cvt_f32_bf16<<<4096, 256, 0, stream>>>(x, xb, T_TOK * H_DIM);
  // attn_w [H][H] -> awT [j][k]
  transpose_cvt_bf16<<<dim3(16, 16, 1), 256, 0, stream>>>(attn_w, awT, H_DIM, H_DIM);
  // w1 [E][H][F] -> w1T [E][F][H]
  transpose_cvt_bf16<<<dim3(F_DIM / 64, H_DIM / 64, E_NUM), 256, 0, stream>>>(
      w1, w1t, H_DIM, F_DIM);
  // w2 [E][F][H] -> w2T [E][H][F]
  transpose_cvt_bf16<<<dim3(H_DIM / 64, F_DIM / 64, E_NUM), 256, 0, stream>>>(
      w2, w2t, F_DIM, H_DIM);
  (void)hipMemsetAsync(d_out, 0, (size_t)T_TOK * H_DIM * sizeof(float), stream);

  attn_gemm<<<dim3(H_DIM / 128, T_TOK / 128), 256, 0, stream>>>(xb, awT, attn_b, hb);
  router<<<T_TOK / 8, 256, 0, stream>>>(hb, router_w, topi, topv);
  build_lists<<<E_NUM, 256, 0, stream>>>(topi, topv, cntb, idxb, gvb);

  size_t lds = (size_t)(32 * 1024 + 32 * 128 + 2 * 128 * 32 + 1024 * 32) * 2; // 155648 B
  moe_expert<<<E_NUM * (T_TOK / 32), 256, lds, stream>>>(hb, w1t, w2t, cntb,
                                                         idxb, gvb, out);
}
